// SparseAttention_19224273617312
// MI455X (gfx1250) — matrix operand order repaired and checked by reading, not compiled, not run on hardware
//
#include <hip/hip_runtime.h>

// ---------------- problem constants ----------------
#define BB 4
#define SS 2048
#define DD 512
#define HH 8
#define DKK 64
#define DVV 64
#define MROWS (BB * SS)          // 8192
#define ALPHA 0.6f
#define EPSR 1e-9f

typedef __bf16 bf16_t;
typedef __bf16 v8bf  __attribute__((ext_vector_type(8)));
typedef __bf16 v16bf __attribute__((ext_vector_type(16)));
typedef float  v4fv  __attribute__((ext_vector_type(4)));
typedef float  v8f   __attribute__((ext_vector_type(8)));

__device__ __forceinline__ v8f wmma_bf16(v16bf a, v16bf b, v8f c) {
    return __builtin_amdgcn_wmma_f32_16x16x32_bf16(
        false, a, false, b, (short)0, c, false, false);
}
__device__ __forceinline__ v16bf cat16(v8bf lo, v8bf hi) {
    return __builtin_shufflevector(lo, hi, 0, 1, 2, 3, 4, 5, 6, 7,
                                   8, 9, 10, 11, 12, 13, 14, 15);
}

// ---------------- workspace layout (bytes) ----------------
#define WS_WQT   ((size_t)0)
#define WS_WKT   (WS_WQT + (size_t)DD * DD * 2)
#define WS_WVT   (WS_WKT + (size_t)DD * DD * 2)
#define WS_WOT   (WS_WVT + (size_t)DD * DD * 2)
#define WS_Q     (WS_WOT + (size_t)DD * DD * 2)              // [B,H,S,DK] bf16
#define WS_K     (WS_Q   + (size_t)MROWS * DD * 2)           // [B,H,S,DK] bf16
#define WS_VT    (WS_K   + (size_t)MROWS * DD * 2)           // [B,H,DV,S] bf16
#define WS_ATTN  (WS_VT  + (size_t)MROWS * DD * 2)           // [B,S,H*DV] bf16
#define WS_XB    (WS_ATTN+ (size_t)MROWS * DD * 2)           // [B*S,D]    bf16

// =====================================================================
// K0a: transpose + convert 512x512 fp32 weights to bf16 W^T[n][k]
// =====================================================================
__global__ void wtrans_kernel(const float* wq, const float* wk,
                              const float* wv, const float* wo,
                              bf16_t* oq, bf16_t* ok_, bf16_t* ov, bf16_t* oo) {
    int idx = blockIdx.x * blockDim.x + threadIdx.x;   // n*512 + k
    int n = idx >> 9, k = idx & 511;
    const float* in; bf16_t* out;
    switch (blockIdx.y) {
        case 0:  in = wq; out = oq;  break;
        case 1:  in = wk; out = ok_; break;
        case 2:  in = wv; out = ov;  break;
        default: in = wo; out = oo;  break;
    }
    out[idx] = (bf16_t)in[k * DD + n];
}

// =====================================================================
// K0b: convert x fp32 -> bf16 (removes cvt from GEMM hot loops)
// =====================================================================
__global__ void xconv_kernel(const float* __restrict__ x, bf16_t* __restrict__ xb) {
    int idx = (blockIdx.x * blockDim.x + threadIdx.x) * 4;
    v4fv v = *(const v4fv*)(x + idx);
    xb[idx + 0] = (bf16_t)v[0];
    xb[idx + 1] = (bf16_t)v[1];
    xb[idx + 2] = (bf16_t)v[2];
    xb[idx + 3] = (bf16_t)v[3];
}

// =====================================================================
// K1/K3: unified GEMM. One wave/block, 32(M) x 64(N) tile, K=512.
// Software-pipelined (ping-pong A/B fragment buffers). launch_bounds
// pins min-waves-per-EU=1 so the allocator can use the full VGPR file
// (prevents the spills observed with the default occupancy target).
// mode = mode_base + blockIdx.z:
//   0 -> q  [B,H,S,DK] * 0.125   1 -> k [B,H,S,DK]
//   2 -> v^T [B,H,DV,S]          3 -> fp32 out [M,D] (output projection)
// =====================================================================
__global__ __launch_bounds__(32, 1)
void gemm_kernel(const bf16_t* __restrict__ Aq,   // xb for modes 0-2
                 const bf16_t* __restrict__ Ao,   // attn for mode 3
                 const bf16_t* __restrict__ wqT,
                 const bf16_t* __restrict__ wkT,
                 const bf16_t* __restrict__ wvT,
                 const bf16_t* __restrict__ woT,
                 const float* __restrict__ bq,
                 const float* __restrict__ bk,
                 const float* __restrict__ bv,
                 const float* __restrict__ bo,
                 bf16_t* __restrict__ qo,
                 bf16_t* __restrict__ ko,
                 bf16_t* __restrict__ vto,
                 float* __restrict__ fout,
                 int mode_base) {
    const int mode = mode_base + blockIdx.z;
    const bf16_t* A   = (mode == 3) ? Ao : Aq;
    const bf16_t* WT  = (mode == 0) ? wqT : (mode == 1) ? wkT
                                          : (mode == 2) ? wvT : woT;
    const float*  bia = (mode == 0) ? bq : (mode == 1) ? bk
                                         : (mode == 2) ? bv : bo;

    const int lane = threadIdx.x & 31;
    const int lm   = lane & 15;
    const int half = lane >> 4;
    const int ms0  = blockIdx.x * 32;
    const int nc0  = blockIdx.y * 64;

    const bf16_t* ar0 = A + (size_t)(ms0 + lm) * DD;
    const bf16_t* ar1 = A + (size_t)(ms0 + 16 + lm) * DD;
    const bf16_t* w0  = WT + (size_t)(nc0 +  0 + lm) * DD;
    const bf16_t* w1  = WT + (size_t)(nc0 + 16 + lm) * DD;
    const bf16_t* w2  = WT + (size_t)(nc0 + 32 + lm) * DD;
    const bf16_t* w3  = WT + (size_t)(nc0 + 48 + lm) * DD;

    v8f acc[2][4];
#pragma unroll
    for (int i = 0; i < 2; ++i)
#pragma unroll
        for (int j = 0; j < 4; ++j) acc[i][j] = (v8f){};

    auto loadAB = [&](int kk, v16bf& a0, v16bf& a1,
                      v16bf& b0, v16bf& b1, v16bf& b2, v16bf& b3) {
        const int k0 = kk * 32 + 16 * half;
        a0 = cat16(*(const v8bf*)(ar0 + k0), *(const v8bf*)(ar0 + k0 + 8));
        a1 = cat16(*(const v8bf*)(ar1 + k0), *(const v8bf*)(ar1 + k0 + 8));
        const int kb = kk * 32 + 16 * half;
        b0 = *(const v16bf*)(w0 + kb);
        b1 = *(const v16bf*)(w1 + kb);
        b2 = *(const v16bf*)(w2 + kb);
        b3 = *(const v16bf*)(w3 + kb);
    };
    auto mma8 = [&](v16bf a0, v16bf a1, v16bf b0, v16bf b1, v16bf b2, v16bf b3) {
        acc[0][0] = wmma_bf16(a0, b0, acc[0][0]);
        acc[1][0] = wmma_bf16(a1, b0, acc[1][0]);
        acc[0][1] = wmma_bf16(a0, b1, acc[0][1]);
        acc[1][1] = wmma_bf16(a1, b1, acc[1][1]);
        acc[0][2] = wmma_bf16(a0, b2, acc[0][2]);
        acc[1][2] = wmma_bf16(a1, b2, acc[1][2]);
        acc[0][3] = wmma_bf16(a0, b3, acc[0][3]);
        acc[1][3] = wmma_bf16(a1, b3, acc[1][3]);
    };

    v16bf a0A, a1A, b0A, b1A, b2A, b3A;
    v16bf a0B, a1B, b0B, b1B, b2B, b3B;
    loadAB(0, a0A, a1A, b0A, b1A, b2A, b3A);
#pragma unroll 2
    for (int kk = 0; kk < 16; kk += 2) {
        loadAB(kk + 1, a0B, a1B, b0B, b1B, b2B, b3B);
        mma8(a0A, a1A, b0A, b1A, b2A, b3A);
        if (kk + 2 < 16) loadAB(kk + 2, a0A, a1A, b0A, b1A, b2A, b3A);
        mma8(a0B, a1B, b0B, b1B, b2B, b3B);
    }

#pragma unroll
    for (int mh = 0; mh < 2; ++mh) {
#pragma unroll
        for (int j = 0; j < 4; ++j) {
#pragma unroll
            for (int r = 0; r < 8; ++r) {
                const int m = ms0 + mh * 16 + r + 8 * half;  // row = b*S + s
                const int n = nc0 + j * 16 + lm;             // col = h*64 + d
                float val = acc[mh][j][r] + bia[n];
                const int b = m >> 11, s = m & (SS - 1);
                const int h = n >> 6,  d = n & 63;
                if (mode == 0) {
                    val *= 0.125f;    // 1/sqrt(DK) folded into q
                    qo[((size_t)(b * HH + h) * SS + s) * DKK + d] = (bf16_t)val;
                } else if (mode == 1) {
                    ko[((size_t)(b * HH + h) * SS + s) * DKK + d] = (bf16_t)val;
                } else if (mode == 2) {
                    vto[((size_t)(b * HH + h) * DVV + d) * SS + s] = (bf16_t)val;
                } else {
                    fout[(size_t)m * DD + n] = val;
                }
            }
        }
    }
}

// =====================================================================
// K2: fused sparse attention. Block = 4 waves; each wave owns 16 q rows.
// LDS: per wave 16 x 2056 bf16 e-row buffer (+ stat arrays).
// =====================================================================
#define ESTR 2056   // padded row stride (elements) to break LDS bank conflicts

__global__ __launch_bounds__(128, 1)
void attn_kernel(const bf16_t* __restrict__ qbuf,
                 const bf16_t* __restrict__ kbuf,
                 const bf16_t* __restrict__ vtbuf,
                 bf16_t* __restrict__ attn_out) {
    extern __shared__ char smem[];
    const int wave = threadIdx.x >> 5;
    const int lane = threadIdx.x & 31;
    const int lm   = lane & 15;
    const int half = lane >> 4;

    const int bh    = blockIdx.y;                 // b*H + h
    const int qbase = blockIdx.x * 64 + wave * 16;

    const bf16_t* qp  = qbuf  + ((size_t)bh * SS + qbase) * DKK;
    const bf16_t* kp  = kbuf  + (size_t)bh * SS * DKK;
    const bf16_t* vtp = vtbuf + (size_t)bh * DVV * SS;

    bf16_t* etile = ((bf16_t*)smem) + (size_t)wave * 16 * ESTR;
    float* rowmax = (float*)(smem + (size_t)4 * 16 * ESTR * 2);
    float* rowC   = rowmax + 64;
    float* rowThr = rowC + 64;

    // ---- Phase 1: scores -> LDS (bf16), per-lane row maxes, pipelined ----
    v16bf aq0, aq1;
    {
        const bf16_t* qr = qp + (size_t)lm * DKK;
        const int k0 = 16 * half;
        aq0 = cat16(*(const v8bf*)(qr + k0),      *(const v8bf*)(qr + k0 + 8));
        aq1 = cat16(*(const v8bf*)(qr + 32 + k0), *(const v8bf*)(qr + 40 + k0));
    }
    float maxv[8];
#pragma unroll
    for (int r = 0; r < 8; ++r) maxv[r] = -1e30f;

    auto loadK = [&](int n0, v16bf& k0v, v16bf& k1v) {
        const bf16_t* kr = kp + (size_t)(n0 + lm) * DKK + 16 * half;
        k0v = *(const v16bf*)(kr);
        k1v = *(const v16bf*)(kr + 32);
    };
    auto score_tile = [&](int n0, v16bf k0v, v16bf k1v) {
        v8f sc = {};
        sc = wmma_bf16(aq0, k0v, sc);
        sc = wmma_bf16(aq1, k1v, sc);
#pragma unroll
        for (int r = 0; r < 8; ++r) {
            float v = sc[r];
            maxv[r] = fmaxf(maxv[r], v);
            etile[(size_t)(r + 8 * half) * ESTR + n0 + lm] = (bf16_t)v;
        }
    };

    v16bf kA0, kA1, kB0, kB1;
    loadK(0, kA0, kA1);
    for (int n0 = 0; n0 < SS; n0 += 32) {
        loadK(n0 + 16, kB0, kB1);
        if (n0 + 128 < SS)
            __builtin_prefetch(kp + (size_t)(n0 + 128 + lm) * DKK, 0, 0);
        score_tile(n0, kA0, kA1);
        if (n0 + 32 < SS) loadK(n0 + 32, kA0, kA1);
        score_tile(n0 + 16, kB0, kB1);
    }
#pragma unroll
    for (int m = 1; m < 16; m <<= 1)
#pragma unroll
        for (int r = 0; r < 8; ++r)
            maxv[r] = fmaxf(maxv[r], __shfl_xor(maxv[r], m, 32));
    if (lm == 0)
#pragma unroll
        for (int r = 0; r < 8; ++r)
            rowmax[wave * 16 + r + 8 * half] = maxv[r];
    __syncthreads();

    // ---- Phase 2: e = exp(s-m), Z, then 3 threshold scans (vectorized) ----
    {
        const int row = lane >> 1;
        const int c0  = (lane & 1) * 8;
        const float m = rowmax[wave * 16 + row];
        bf16_t* er = etile + (size_t)row * ESTR;
        float z = 0.f;
        for (int c = c0; c < SS; c += 16) {
            v8bf ev = *(const v8bf*)(er + c);
            v8bf eo;
#pragma unroll
            for (int i = 0; i < 8; ++i) {
                float e = __expf((float)ev[i] - m);
                z += e;
                eo[i] = (bf16_t)e;
            }
            *(v8bf*)(er + c) = eo;
        }
        z += __shfl_xor(z, 1, 32);
        float C = z, tmax = 0.f;
        const float thrf[3] = {1.f / (float)SS,
                               1.f / (ALPHA * (float)SS),
                               1.f / (ALPHA * ALPHA * (float)SS)};
#pragma unroll
        for (int t = 0; t < 3; ++t) {
            tmax = fmaxf(tmax, C * thrf[t]);
            float ssum = 0.f;
            for (int c = c0; c < SS; c += 16) {
                v8bf ev = *(const v8bf*)(er + c);
#pragma unroll
                for (int i = 0; i < 8; ++i) {
                    float e = (float)ev[i];
                    ssum += (e >= tmax) ? e : 0.f;
                }
            }
            ssum += __shfl_xor(ssum, 1, 32);
            C = ssum + EPSR * C;
        }
        if (c0 == 0) {
            rowC[wave * 16 + row]   = C;
            rowThr[wave * 16 + row] = tmax;
        }
    }
    __syncthreads();

    // ---- Phase 3: out = (masked e / C) @ V^T via WMMA, pipelined ----
    const float tm = rowThr[wave * 16 + lm];     // A-operand row threshold
    float cin[8];
#pragma unroll
    for (int r = 0; r < 8; ++r)
        cin[r] = 1.0f / rowC[wave * 16 + r + 8 * half];

    v8f acc[4];
#pragma unroll
    for (int j = 0; j < 4; ++j) acc[j] = (v8f){};

    const bf16_t* erow = etile + (size_t)lm * ESTR;
    auto loadE = [&](int kb, v8bf& e0, v8bf& e1) {
        const int k0 = kb * 32 + 16 * half;
        e0 = *(const v8bf*)(erow + k0);
        e1 = *(const v8bf*)(erow + k0 + 8);
    };
    auto loadV = [&](int kb, v16bf v[4]) {
        const int kv = kb * 32 + 16 * half;
        v[0] = *(const v16bf*)(vtp + (size_t)( 0 + lm) * SS + kv);
        v[1] = *(const v16bf*)(vtp + (size_t)(16 + lm) * SS + kv);
        v[2] = *(const v16bf*)(vtp + (size_t)(32 + lm) * SS + kv);
        v[3] = *(const v16bf*)(vtp + (size_t)(48 + lm) * SS + kv);
    };
    auto attend = [&](v8bf e0, v8bf e1, v16bf v[4]) {
        v16bf af;
#pragma unroll
        for (int i = 0; i < 8; ++i) {
            af[i]     = ((float)e0[i] >= tm) ? e0[i] : (bf16_t)0.f;
            af[8 + i] = ((float)e1[i] >= tm) ? e1[i] : (bf16_t)0.f;
        }
        acc[0] = wmma_bf16(af, v[0], acc[0]);
        acc[1] = wmma_bf16(af, v[1], acc[1]);
        acc[2] = wmma_bf16(af, v[2], acc[2]);
        acc[3] = wmma_bf16(af, v[3], acc[3]);
    };

    v8bf eA0, eA1, eB0, eB1;
    v16bf vA[4], vB[4];
    loadE(0, eA0, eA1);
    loadV(0, vA);
    for (int kb = 0; kb < SS / 32; kb += 2) {
        loadE(kb + 1, eB0, eB1);
        loadV(kb + 1, vB);
        if (kb + 4 < SS / 32)
            __builtin_prefetch(vtp + (size_t)lm * SS + (kb + 4) * 32, 0, 0);
        attend(eA0, eA1, vA);
        if (kb + 2 < SS / 32) {
            loadE(kb + 2, eA0, eA1);
            loadV(kb + 2, vA);
        }
        attend(eB0, eB1, vB);
    }

    const int b = bh / HH, h = bh % HH;
#pragma unroll
    for (int j = 0; j < 4; ++j) {
#pragma unroll
        for (int r = 0; r < 8; ++r) {
            const int qg = qbase + r + 8 * half;
            size_t off = ((size_t)b * SS + qg) * (HH * DVV) + h * DVV + j * 16 + lm;
            attn_out[off] = (bf16_t)(acc[j][r] * cin[r]);
        }
    }
}

// =====================================================================
extern "C" void kernel_launch(void* const* d_in, const int* in_sizes, int n_in,
                              void* d_out, int out_size, void* d_ws, size_t ws_size,
                              hipStream_t stream) {
    const float* x  = (const float*)d_in[0];
    const float* Wq = (const float*)d_in[1];
    const float* bq = (const float*)d_in[2];
    const float* Wk = (const float*)d_in[3];
    const float* bk = (const float*)d_in[4];
    const float* Wv = (const float*)d_in[5];
    const float* bv = (const float*)d_in[6];
    const float* Wo = (const float*)d_in[7];
    const float* bo = (const float*)d_in[8];
    float* out = (float*)d_out;

    char* ws = (char*)d_ws;
    bf16_t* wqT  = (bf16_t*)(ws + WS_WQT);
    bf16_t* wkT  = (bf16_t*)(ws + WS_WKT);
    bf16_t* wvT  = (bf16_t*)(ws + WS_WVT);
    bf16_t* woT  = (bf16_t*)(ws + WS_WOT);
    bf16_t* qb   = (bf16_t*)(ws + WS_Q);
    bf16_t* kb   = (bf16_t*)(ws + WS_K);
    bf16_t* vtb  = (bf16_t*)(ws + WS_VT);
    bf16_t* attn = (bf16_t*)(ws + WS_ATTN);
    bf16_t* xb   = (bf16_t*)(ws + WS_XB);

    // K0a: weight transpose + bf16 convert
    wtrans_kernel<<<dim3(DD * DD / 256, 4), 256, 0, stream>>>(
        Wq, Wk, Wv, Wo, wqT, wkT, wvT, woT);

    // K0b: x -> bf16
    xconv_kernel<<<dim3((size_t)MROWS * DD / 4 / 256), 256, 0, stream>>>(x, xb);

    // K1: QKV projections (z: 0=q scaled, 1=k, 2=v transposed)
    gemm_kernel<<<dim3(MROWS / 32, DD / 64, 3), 32, 0, stream>>>(
        xb, attn, wqT, wkT, wvT, woT, bq, bk, bv, bo, qb, kb, vtb, out, 0);

    // K2: fused sparse attention (4 waves/block, 64 q-rows/block)
    const size_t smem = (size_t)4 * 16 * ESTR * 2 + 3 * 64 * sizeof(float);
    attn_kernel<<<dim3(SS / 64, BB * HH), 128, smem, stream>>>(qb, kb, vtb, attn);

    // K3: output projection (mode 3 -> fp32 d_out)
    gemm_kernel<<<dim3(MROWS / 32, DD / 64, 1), 32, 0, stream>>>(
        xb, attn, wqT, wkT, wvT, woT, bq, bk, bv, bo, qb, kb, vtb, out, 3);
}
